// KHopGTModel_8143257994121
// MI455X (gfx1250) — matrix operand order repaired and checked by reading, not compiled, not run on hardware
//
#include <hip/hip_runtime.h>
#include <hip/hip_bf16.h>

typedef __attribute__((ext_vector_type(16))) __bf16 v16bf;
typedef __attribute__((ext_vector_type(8)))  __bf16 v8bf;
typedef __attribute__((ext_vector_type(8)))  float  v8f;

#if __has_builtin(__builtin_amdgcn_sched_barrier)
#define SCHED_FENCE() __builtin_amdgcn_sched_barrier(0)
#else
#define SCHED_FENCE()
#endif

// ---------------------------------------------------------------------------
// Wave helpers (wave32)
// ---------------------------------------------------------------------------
__device__ __forceinline__ float wave_sum32(float v) {
#pragma unroll
  for (int o = 16; o > 0; o >>= 1) v += __shfl_xor(v, o, 32);
  return v;
}

// monotonic float<->u32 key for atomicMax on floats (incl. negatives)
__device__ __forceinline__ unsigned f2key(float f) {
  int b = __float_as_int(f);
  return (unsigned)(b ^ ((b >> 31) | 0x80000000));
}
__device__ __forceinline__ float key2f(unsigned k) {
  int b = (k & 0x80000000u) ? (int)(k ^ 0x80000000u) : ~(int)k;
  return __int_as_float(b);
}

// ---------------------------------------------------------------------------
// Weight prep: W[l][k][n] fp32 -> WT[mat][n*128+k] bf16 (transposed, K contiguous)
// mat order: l0{q,k,v,o}, l1{q,k,v,o}
// ---------------------------------------------------------------------------
__global__ void prep_weights(const float* __restrict__ Wq, const float* __restrict__ Wk,
                             const float* __restrict__ Wv, const float* __restrict__ Wo,
                             __bf16* __restrict__ WT) {
  int idx = blockIdx.x * blockDim.x + threadIdx.x;
  if (idx >= 8 * 128 * 128) return;
  int mat = idx >> 14;
  int e   = idx & 16383;
  int n = e >> 7, k = e & 127;
  int l = mat >> 2, which = mat & 3;
  const float* W = (which == 0) ? Wq : (which == 1) ? Wk : (which == 2) ? Wv : Wo;
  WT[(size_t)mat * 16384 + n * 128 + k] = (__bf16)W[(size_t)l * 16384 + k * 128 + n];
}

__global__ void f32_to_bf16(const float* __restrict__ in, __bf16* __restrict__ out, int n) {
  int i = blockIdx.x * blockDim.x + threadIdx.x;
  if (i < n) out[i] = (__bf16)in[i];
}

// ---------------------------------------------------------------------------
// GEMM building blocks. Matrix layouts used here (A/B k order is contiguous, not the ISA table order):
//  A 16x32 bf16: lane half hh -> K chunk [16hh..16hh+15]
//  B 32x16 bf16: col = lane&15, K base = 16*hh (consumed from transposed WT[n][k])
//  C/D f32:      VGPR rr -> row rr + 8*hh, col = lane&15
// ---------------------------------------------------------------------------
__device__ __forceinline__ void load_a_tiles(const __bf16* __restrict__ Arow, int hh,
                                             v16bf a[4]) {
#pragma unroll
  for (int kk = 0; kk < 4; ++kk) {
    int kb = kk * 32;
    v8bf lo = *(const v8bf*)(Arow + kb + 16 * hh);
    v8bf hi = *(const v8bf*)(Arow + kb + 16 * hh + 8);
    a[kk] = __builtin_shufflevector(lo, hi, 0, 1, 2, 3, 4, 5, 6, 7, 8, 9, 10, 11, 12, 13, 14, 15);
  }
}

// 4-tile (16 rows x 64 cols) sweep with explicit double buffering. A scheduling
// fence at the end of each tile stops the compiler from sinking stores /
// collapsing the B double-buffer (which previously serialized every WMMA on
// s_wait_loadcnt 0x0 and blew up accumulator pressure).
template <bool FULL>
__device__ __forceinline__ void gemm_tiles4(const v16bf a[4],
                                            const __bf16* __restrict__ WTm,
                                            const float* __restrict__ bias,
                                            float* __restrict__ Out,
                                            int row0, int ctBase, int r, int hh, int nrows) {
  const __bf16* Bbase = WTm + (size_t)(ctBase * 16 + r) * 128 + 16 * hh;
  v16bf bcur[4], bnxt[4];
#pragma unroll
  for (int kk = 0; kk < 4; ++kk) bcur[kk] = *(const v16bf*)(Bbase + kk * 32);

#pragma unroll
  for (int i = 0; i < 4; ++i) {
    if (i < 3) {
      const __bf16* Bn = Bbase + (size_t)(i + 1) * (16 * 128);
#pragma unroll
      for (int kk = 0; kk < 4; ++kk) bnxt[kk] = *(const v16bf*)(Bn + kk * 32);
    }
    v8f acc = {};
#pragma unroll
    for (int kk = 0; kk < 4; ++kk) {
      acc = __builtin_amdgcn_wmma_f32_16x16x32_bf16(false, a[kk], false, bcur[kk],
                                                    (short)0, acc, false, false);
    }
    int ct = ctBase + i;
    float bv_ = bias[ct * 16 + r];
    if (FULL) {
      float* outBase = Out + (size_t)(row0 + 8 * hh) * 128 + ct * 16 + r;
#pragma unroll
      for (int rr = 0; rr < 8; ++rr) outBase[(size_t)rr * 128] = acc[rr] + bv_;
    } else {
#pragma unroll
      for (int rr = 0; rr < 8; ++rr) {
        int m = row0 + rr + 8 * hh;
        if (m < nrows) Out[(size_t)m * 128 + ct * 16 + r] = acc[rr] + bv_;
      }
    }
#pragma unroll
    for (int kk = 0; kk < 4; ++kk) bcur[kk] = bnxt[kk];   // renamed away by unroll
    SCHED_FENCE();   // keep per-tile pipeline: next tile's loads already in flight
  }
}

// ---------------------------------------------------------------------------
// Projection GEMM: grid.x -> 16-row blocks, grid.y -> matrix (Q/K/V or just O),
// grid.z -> column half. Small work per wave = many waves = latency tolerance
// (total GEMM FLOPs are tiny; the win is filling every SIMD on the chip).
// ---------------------------------------------------------------------------
struct ProjArgs {
  const __bf16* wt;
  const float*  bias;
  float*        out;
};

__global__ void __launch_bounds__(256) gemm_proj(const __bf16* __restrict__ A,
                                                 ProjArgs p0, ProjArgs p1, ProjArgs p2,
                                                 int nrows) {
  int lane = threadIdx.x & 31;
  int wid  = (blockIdx.x * blockDim.x + threadIdx.x) >> 5;
  int row0 = wid * 16;
  if (row0 >= nrows) return;                 // wave-uniform exit (EXEC all-1 for WMMA)
  ProjArgs p = (blockIdx.y == 0) ? p0 : (blockIdx.y == 1) ? p1 : p2;
  int ctBase = blockIdx.z * 4;               // column half: tiles [ctBase, ctBase+4)
  int r  = lane & 15;
  int hh = lane >> 4;
  int arow = row0 + r; if (arow >= nrows) arow = nrows - 1;

  v16bf a[4];
  load_a_tiles(A + (size_t)arow * 128, hh, a);

  if (row0 + 16 <= nrows) {                  // uniform branch, hoisted
    gemm_tiles4<true>(a, p.wt, p.bias, p.out, row0, ctBase, r, hh, nrows);
  } else {
    gemm_tiles4<false>(a, p.wt, p.bias, p.out, row0, ctBase, r, hh, nrows);
  }
}

// ---------------------------------------------------------------------------
// Edge pass 1: per-edge per-head logits + segment max (keyed by src)
// one wave per edge; lane handles 4 contiguous channels (one head = 4 lanes)
// ---------------------------------------------------------------------------
__global__ void edge_logits_max(const int* __restrict__ src, const int* __restrict__ dst,
                                const float* __restrict__ Q, const float* __restrict__ K,
                                float* __restrict__ ex, unsigned* __restrict__ Mkey, int nE) {
  int lane = threadIdx.x & 31;
  int e = (blockIdx.x * blockDim.x + threadIdx.x) >> 5;
  if (e >= nE) return;
  int s = src[e], d = dst[e];
  float4 q = *(const float4*)(Q + (size_t)s * 128 + lane * 4);
  float4 k = *(const float4*)(K + (size_t)d * 128 + lane * 4);
  float p = q.x * k.x + q.y * k.y + q.z * k.z + q.w * k.w;
  p += __shfl_xor(p, 1, 32);
  p += __shfl_xor(p, 2, 32);                 // groups of 4 lanes = one head (HD=16)
  if ((lane & 3) == 0) {
    int hh = lane >> 2;
    float logit = p * 0.25f;                 // 1/sqrt(HD)
    ex[(size_t)e * 8 + hh] = logit;
    atomicMax(&Mkey[s * 8 + hh], f2key(logit));
  }
}

// ---------------------------------------------------------------------------
// Edge pass 2: ex = exp(logit - m); den[src,h] += ex
// ---------------------------------------------------------------------------
__global__ void edge_exp_den(const int* __restrict__ src, const unsigned* __restrict__ Mkey,
                             float* __restrict__ ex, float* __restrict__ den, int nEH) {
  int idx = blockIdx.x * blockDim.x + threadIdx.x;
  if (idx >= nEH) return;
  int e = idx >> 3, hh = idx & 7;
  int s = src[e];
  float m = key2f(Mkey[s * 8 + hh]);
  float v = __expf(ex[idx] - m);
  ex[idx] = v;
  atomicAdd(&den[s * 8 + hh], v);
}

// ---------------------------------------------------------------------------
// Edge pass 3: agg[src] += (ex/den) * V[dst]   (one wave per edge, 4 ch/lane)
// ---------------------------------------------------------------------------
__global__ void edge_aggregate(const int* __restrict__ src, const int* __restrict__ dst,
                               const float* __restrict__ ex, const float* __restrict__ den,
                               const float* __restrict__ V, float* __restrict__ agg, int nE) {
  int lane = threadIdx.x & 31;
  int e = (blockIdx.x * blockDim.x + threadIdx.x) >> 5;
  if (e >= nE) return;
  int s = src[e], d = dst[e];
  int hh = lane >> 2;
  float w = ex[(size_t)e * 8 + hh] / (den[s * 8 + hh] + 1e-16f);
  float4 v = *(const float4*)(V + (size_t)d * 128 + lane * 4);
  float* ag = agg + (size_t)s * 128 + lane * 4;
  atomicAdd(ag + 0, w * v.x);
  atomicAdd(ag + 1, w * v.y);
  atomicAdd(ag + 2, w * v.z);
  atomicAdd(ag + 3, w * v.w);
}

// ---------------------------------------------------------------------------
// LayerNorm over 128 channels, one wave per node; emits bf16 (and optional fp32)
// ---------------------------------------------------------------------------
__global__ void layernorm_wave(const float* __restrict__ X, const float* __restrict__ g,
                               const float* __restrict__ be, __bf16* __restrict__ Yb,
                               float* __restrict__ Yf, int nrows) {
  int lane = threadIdx.x & 31;
  int n = (blockIdx.x * blockDim.x + threadIdx.x) >> 5;
  if (n >= nrows) return;
  float4 x = *(const float4*)(X + (size_t)n * 128 + lane * 4);
  float s  = x.x + x.y + x.z + x.w;
  float ss = x.x * x.x + x.y * x.y + x.z * x.z + x.w * x.w;
  s = wave_sum32(s);
  ss = wave_sum32(ss);
  float mu  = s * (1.0f / 128.0f);
  float var = ss * (1.0f / 128.0f) - mu * mu;
  float rs  = rsqrtf(var + 1e-5f);
  float xv[4] = {x.x, x.y, x.z, x.w};
#pragma unroll
  for (int j = 0; j < 4; ++j) {
    int c = lane * 4 + j;
    float y = (xv[j] - mu) * rs * g[c] + be[c];
    Yb[(size_t)n * 128 + c] = (__bf16)y;
    if (Yf) Yf[(size_t)n * 128 + c] = y;
  }
}

// ---------------------------------------------------------------------------
// Init per-layer reduction buffers
// ---------------------------------------------------------------------------
__global__ void init_buffers(unsigned* __restrict__ Mkey, float* __restrict__ den,
                             float* __restrict__ agg, int nN) {
  int idx = blockIdx.x * blockDim.x + threadIdx.x;
  if (idx < nN * 128) agg[idx] = 0.0f;
  if (idx < nN * 8) { Mkey[idx] = 0u; den[idx] = 0.0f; }
}

// ---------------------------------------------------------------------------
extern "C" void kernel_launch(void* const* d_in, const int* in_sizes, int n_in,
                              void* d_out, int out_size, void* d_ws, size_t ws_size,
                              hipStream_t stream) {
  const float* x   = (const float*)d_in[0];
  const int*   ei  = (const int*)d_in[1];
  const float* Wq  = (const float*)d_in[2];
  const float* bq  = (const float*)d_in[3];
  const float* Wk  = (const float*)d_in[4];
  const float* bk  = (const float*)d_in[5];
  const float* Wv  = (const float*)d_in[6];
  const float* bv  = (const float*)d_in[7];
  const float* Wo  = (const float*)d_in[8];
  const float* bo  = (const float*)d_in[9];
  const float* g1  = (const float*)d_in[10];
  const float* be1 = (const float*)d_in[11];
  const float* g2  = (const float*)d_in[12];
  const float* be2 = (const float*)d_in[13];

  const int N  = in_sizes[0] / 128;
  const int En = in_sizes[1] / 2;
  const int* srcI = ei;
  const int* dstI = ei + En;

  // ---- workspace carve (256B aligned) ----
  size_t off = 0;
  auto carve = [&](size_t bytes) -> char* {
    char* p = (char*)d_ws + off;
    off += (bytes + 255) & ~(size_t)255;
    return p;
  };
  __bf16*   WT   = (__bf16*)carve(8 * 16384 * sizeof(__bf16));
  __bf16*   hb   = (__bf16*)carve((size_t)N * 128 * sizeof(__bf16));
  float*    Qb   = (float*)carve((size_t)N * 128 * sizeof(float));
  float*    Kb   = (float*)carve((size_t)N * 128 * sizeof(float));
  float*    Vb   = (float*)carve((size_t)N * 128 * sizeof(float));
  float*    ex   = (float*)carve((size_t)En * 8 * sizeof(float));
  unsigned* Mkey = (unsigned*)carve((size_t)N * 8 * sizeof(unsigned));
  float*    den  = (float*)carve((size_t)N * 8 * sizeof(float));
  float*    agg  = (float*)carve((size_t)N * 128 * sizeof(float));
  // aliases: Q dead after edge pass 1, K dead after edge pass 1
  __bf16* aggn = (__bf16*)Qb;   // bf16 LN1 output (GEMM-O input)
  float*  outb = (float*)Kb;    // O-projection output (LN2 input)

  const int B = 256;
  // --- one-time prep (recomputed every call: deterministic) ---
  prep_weights<<<(8 * 16384 + B - 1) / B, B, 0, stream>>>(Wq, Wk, Wv, Wo, WT);
  f32_to_bf16<<<((size_t)N * 128 + B - 1) / B, B, 0, stream>>>(x, hb, N * 128);

  const int waves16   = (N + 15) / 16;           // GEMM row blocks (16 rows each)
  const int gemmGridX = (waves16 + 7) / 8;       // 8 waves / 256-thread block
  const int edgeGrid  = (En + 7) / 8;            // 1 wave / edge
  const int nodeGrid  = (N + 7) / 8;             // 1 wave / node
  const int ehGrid    = (En * 8 + B - 1) / B;
  const int initGrid  = ((size_t)N * 128 + B - 1) / B;

  for (int l = 0; l < 2; ++l) {
    const __bf16* WTl = WT + (size_t)(l * 4) * 16384;       // q,k,v contiguous
    const __bf16* WTo = WT + (size_t)(l * 4 + 3) * 16384;

    // Q/K/V projections: y = matrix, z = column half -> 6x the waves for latency
    ProjArgs pq{WTl,               bq + l * 128, Qb};
    ProjArgs pk{WTl + 1 * 16384,   bk + l * 128, Kb};
    ProjArgs pv{WTl + 2 * 16384,   bv + l * 128, Vb};
    gemm_proj<<<dim3(gemmGridX, 3, 2), B, 0, stream>>>(hb, pq, pk, pv, N);

    // segment softmax + aggregation over edges
    init_buffers<<<initGrid, B, 0, stream>>>(Mkey, den, agg, N);
    edge_logits_max<<<edgeGrid, B, 0, stream>>>(srcI, dstI, Qb, Kb, ex, Mkey, En);
    edge_exp_den<<<ehGrid, B, 0, stream>>>(srcI, Mkey, ex, den, En * 8);
    edge_aggregate<<<edgeGrid, B, 0, stream>>>(srcI, dstI, ex, den, Vb, agg, En);

    // LN1 -> bf16, O-projection, LN2 -> (bf16 for next layer, fp32 to d_out)
    layernorm_wave<<<nodeGrid, B, 0, stream>>>(agg, g1 + l * 128, be1 + l * 128,
                                               aggn, nullptr, N);
    ProjArgs po{WTo, bo + l * 128, outb};
    gemm_proj<<<dim3(gemmGridX, 1, 2), B, 0, stream>>>(aggn, po, po, po, N);
    layernorm_wave<<<nodeGrid, B, 0, stream>>>(outb, g2 + l * 128, be2 + l * 128,
                                               hb, (float*)d_out, N);
  }
}
